// CrossAttention_49125835932108
// MI455X (gfx1250) — matrix operand order repaired and checked by reading, not compiled, not run on hardware
//
#include <hip/hip_runtime.h>
#include <hip/hip_bf16.h>

// CDNA5 / gfx1250 wave32 WMMA implementation of the cross-attention block.
// Big GEMMs (QK^T and attn@y) use v_wmma_f32_16x16x32_f16 (f16 in, f32 acc).
// Both GEMMs consume their B operand as contiguous per-lane 32B loads by
// keeping K (naturally) and y (via a one-shot global transpose) in
// column-contiguous f16 layout.

typedef __attribute__((ext_vector_type(16))) _Float16 v16h;
typedef __attribute__((ext_vector_type(8)))  float    v8f;
typedef __attribute__((ext_vector_type(4)))  unsigned int u32x4;

union F16Frag {
    v16h  h;
    u32x4 q[2];
};

#define HH 1024
#define WW 1024
#define BB 4
#define CC 8

// ---------------------------------------------------------------------------
// Kernel 1: Q = x . wq + bq ; K = y . wk + bk   (C=8 reduction per pixel)
// ---------------------------------------------------------------------------
__global__ __launch_bounds__(256) void qk_proj_kernel(
    const float* __restrict__ x, const float* __restrict__ y,
    const float* __restrict__ wq, const float* __restrict__ bq,
    const float* __restrict__ wk, const float* __restrict__ bk,
    _Float16* __restrict__ Qh, _Float16* __restrict__ Kh)
{
    const size_t HW = (size_t)HH * WW;
    size_t pix = (size_t)blockIdx.x * blockDim.x + threadIdx.x;   // B*H*W threads
    size_t b   = pix >> 20;
    size_t hw  = pix & (HW - 1);
    const float* xb = x + b * CC * HW + hw;
    const float* yb = y + b * CC * HW + hw;
    float q = bq[0], k = bk[0];
#pragma unroll
    for (int c = 0; c < CC; ++c) {
        q += xb[c * HW] * wq[c];
        k += yb[c * HW] * wk[c];
    }
    Qh[pix] = (_Float16)q;
    Kh[pix] = (_Float16)k;
}

// ---------------------------------------------------------------------------
// Kernel 1b: yt[b,c,w,h] = (f16) y[b,c,h,w]  — one-shot tiled transpose so the
// attn@y GEMM's B fragments become contiguous per-lane loads (no LDS staging
// in the hot loop). Memory-bound: 128 MB read + 64 MB write.
// ---------------------------------------------------------------------------
__global__ __launch_bounds__(256) void transpose_kernel(
    const float* __restrict__ y, _Float16* __restrict__ yt)
{
    __shared__ _Float16 t[32][33];              // +1 pad: no bank conflicts

    const int blk = blockIdx.x;                 // B*C * 32 * 32 tiles
    const int tw  = blk & 31;
    const int th  = (blk >> 5) & 31;
    const int bc  = blk >> 10;                  // fused (b,c) 0..31

    const int tid  = threadIdx.x;
    const int col  = tid & 31;
    const int rowb = tid >> 5;                  // 0..7

    const float* src = y + ((size_t)bc * HH + th * 32) * WW + tw * 32;
#pragma unroll
    for (int i = 0; i < 4; ++i) {
        const int r = rowb + 8 * i;
        t[r][col] = (_Float16)src[(size_t)r * WW + col];   // coalesced along w
    }
    __syncthreads();

    _Float16* dst = yt + ((size_t)bc * WW + tw * 32) * HH + th * 32;
#pragma unroll
    for (int i = 0; i < 4; ++i) {
        const int r = rowb + 8 * i;                        // w-index within tile
        dst[(size_t)r * HH + col] = t[col][r];             // coalesced along h
    }
}

// ---------------------------------------------------------------------------
// Kernel 2: scores = Q K^T / sqrt(W); attn = softmax(scores)  -> f16
// One block = (batch b, 16-row tile). 8 waves, each wave owns 8 column tiles.
// waves_per_eu(4) raises the VGPR budget to the block's occupancy floor
// (4 waves/SIMD in CU mode) so multiple B fragments can stay in flight.
// ---------------------------------------------------------------------------
__global__ __launch_bounds__(256, 1) __attribute__((amdgpu_waves_per_eu(4)))
void attn_kernel(
    const _Float16* __restrict__ Qh, const _Float16* __restrict__ Kh,
    _Float16* __restrict__ attn)
{
    __shared__ float lds_s[16 * HH];            // 64 KB: 16 rows of raw scores

    const int b  = blockIdx.x >> 6;             // H/16 = 64 row tiles per batch
    const int i0 = (blockIdx.x & 63) * 16;

    const _Float16* Qb = Qh + (size_t)b * HH * WW;
    const _Float16* Kb = Kh + (size_t)b * HH * WW;

    const int tid  = threadIdx.x;
    const int wave = tid >> 5;
    const int lane = tid & 31;
    const int lm   = lane & 15;                 // row (A) / col (B) within tile
    const int sel  = lane >> 4;                 // K-half selector

    v8f acc[8] = {};                            // 8 column tiles per wave

    const _Float16* qrowb = Qb + (size_t)(i0 + lm) * WW + 16 * sel;
    const _Float16* krowb = Kb + (size_t)(wave * 8 * 16 + lm) * WW + 16 * sel;

    for (int kw = 0; kw < WW; kw += 32) {
        if (kw + 32 < WW) __builtin_prefetch(krowb + kw + 32, 0, 1);

        // A fragment: Q rows i0..i0+15, K-slice [kw, kw+32)
        F16Frag a;
        a.q[0] = *(const u32x4*)(qrowb + kw);
        a.q[1] = *(const u32x4*)(qrowb + kw + 8);

        // Batch-issue all 8 B-fragment loads (B = K^T: contiguous K rows).
        F16Frag bf[8];
#pragma unroll
        for (int t = 0; t < 8; ++t) {
            const u32x4* kp = (const u32x4*)(krowb + (size_t)t * 16 * WW + kw);
            bf[t].q[0] = kp[0];
            bf[t].q[1] = kp[1];
        }
#pragma unroll
        for (int t = 0; t < 8; ++t)
            acc[t] = __builtin_amdgcn_wmma_f32_16x16x32_f16(
                false, a.h, false, bf[t].h, (short)0, acc[t], false, false);
    }

    // Scatter scores to LDS per C/D layout: VGPR r -> row r+8*sel, col lane&15
#pragma unroll
    for (int t = 0; t < 8; ++t) {
        const int j = (wave * 8 + t) * 16 + lm;
#pragma unroll
        for (int r = 0; r < 8; ++r)
            lds_s[(r + 8 * sel) * HH + j] = acc[t][r];
    }
    __syncthreads();

    // Row softmax: 16 threads per row, exclusive strided ownership.
    const int   row   = tid >> 4;
    const int   p     = tid & 15;
    const float scale = 1.0f / 32.0f;           // 1/sqrt(1024)

    float m = -3.0e38f;
    for (int k = p; k < HH; k += 16) m = fmaxf(m, lds_s[row * HH + k]);
#pragma unroll
    for (int off = 8; off >= 1; off >>= 1) m = fmaxf(m, __shfl_xor(m, off, 16));
    const float ms = m * scale;

    float sum = 0.0f;
    for (int k = p; k < HH; k += 16) {
        float e = __expf(lds_s[row * HH + k] * scale - ms);
        lds_s[row * HH + k] = e;
        sum += e;
    }
#pragma unroll
    for (int off = 8; off >= 1; off >>= 1) sum += __shfl_xor(sum, off, 16);
    const float inv = 1.0f / sum;

    _Float16* arow = attn + ((size_t)b * HH + i0 + row) * HH;
    for (int k = p; k < HH; k += 16) arow[k] = (_Float16)(lds_s[row * HH + k] * inv);
}

// ---------------------------------------------------------------------------
// Kernel 3: T[c'] = attn @ y[c']  (WMMA, B from pre-transposed yt),
//           out = gamma*(wv.T + bv) + x
// Block = (b, 32 rows, 64 cols). 8 waves = 8 value channels, 2 M-tiles each.
// Pure global_load + wmma hot loop; LDS used only once for the channel mix.
// ---------------------------------------------------------------------------
__global__ __launch_bounds__(256, 1) __attribute__((amdgpu_waves_per_eu(4)))
void out_kernel(
    const _Float16* __restrict__ attn, const _Float16* __restrict__ yt,
    const float* __restrict__ x, const float* __restrict__ wv,
    const float* __restrict__ bv, const float* __restrict__ gamma,
    float* __restrict__ out)
{
    __shared__ float tb[CC][32][64];            // 64 KB T tiles for channel mix

    const int blk = blockIdx.x;                 // B * 32 * 16 blocks
    const int wt  = blk & 15;
    const int it  = (blk >> 4) & 31;
    const int b   = blk >> 9;
    const int i0  = it * 32;
    const int w0  = wt * 64;

    const int tid  = threadIdx.x;
    const int wave = tid >> 5;                  // = value channel c
    const int lane = tid & 31;
    const int lm   = lane & 15;
    const int sel  = lane >> 4;
    const int c    = wave;

    // A rows: attn rows i0 + mi*16 + lm
    const _Float16* arow0 = attn + ((size_t)b * HH + i0 + lm) * HH + 16 * sel;
    // B rows: yt row (w-column) w0 + nt*16 + lm of channel c
    const _Float16* brow0 =
        yt + (((size_t)(b * CC + c)) * WW + w0 + lm) * HH + 16 * sel;

    v8f acc[2][4] = {};                         // 2 M-tiles x 4 N-tiles

    for (int j = 0; j < HH; j += 32) {
        if (j + 32 < HH) __builtin_prefetch(brow0 + j + 32, 0, 1);

        // A fragments (shared across the 4 N-tiles)
        F16Frag a[2];
#pragma unroll
        for (int mi = 0; mi < 2; ++mi) {
            const _Float16* ar = arow0 + (size_t)mi * 16 * HH + j;
            a[mi].q[0] = *(const u32x4*)(ar);
            a[mi].q[1] = *(const u32x4*)(ar + 8);
        }
        // B fragments: contiguous 32B per lane from yt (shared across M-tiles)
        F16Frag bf[4];
#pragma unroll
        for (int nt = 0; nt < 4; ++nt) {
            const u32x4* bp = (const u32x4*)(brow0 + (size_t)nt * 16 * HH + j);
            bf[nt].q[0] = bp[0];
            bf[nt].q[1] = bp[1];
        }
#pragma unroll
        for (int mi = 0; mi < 2; ++mi)
#pragma unroll
            for (int nt = 0; nt < 4; ++nt)
                acc[mi][nt] = __builtin_amdgcn_wmma_f32_16x16x32_f16(
                    false, a[mi].h, false, bf[nt].h, (short)0, acc[mi][nt],
                    false, false);
    }

    // T tiles -> LDS for cross-channel mixing
#pragma unroll
    for (int mi = 0; mi < 2; ++mi)
#pragma unroll
        for (int nt = 0; nt < 4; ++nt)
#pragma unroll
            for (int r = 0; r < 8; ++r)
                tb[c][mi * 16 + r + 8 * sel][nt * 16 + lm] = acc[mi][nt][r];
    __syncthreads();

    // out[c] = gamma * (sum_c' wv[c,c'] T[c'] + bv[c]) + x   (softmax rows sum
    // to 1, so bv folds out of the GEMM exactly)
    const float g = gamma[0];
    for (int idx = tid; idx < CC * 32 * 64; idx += 256) {
        const int cc  = idx >> 11;
        const int rem = idx & 2047;
        const int r   = rem >> 6;
        const int ww2 = rem & 63;
        float v = bv[cc];
#pragma unroll
        for (int c2 = 0; c2 < CC; ++c2) v += wv[cc * CC + c2] * tb[c2][r][ww2];
        const size_t o = (((size_t)b * CC + cc) * HH + (i0 + r)) * WW + (w0 + ww2);
        out[o] = g * v + x[o];
    }
}

// ---------------------------------------------------------------------------
extern "C" void kernel_launch(void* const* d_in, const int* in_sizes, int n_in,
                              void* d_out, int out_size, void* d_ws, size_t ws_size,
                              hipStream_t stream) {
    const float* x     = (const float*)d_in[0];
    const float* y     = (const float*)d_in[1];
    const float* wq    = (const float*)d_in[2];
    const float* bq    = (const float*)d_in[3];
    const float* wk    = (const float*)d_in[4];
    const float* bk    = (const float*)d_in[5];
    const float* wv    = (const float*)d_in[6];
    const float* bv    = (const float*)d_in[7];
    const float* gamma = (const float*)d_in[8];
    float* outp = (float*)d_out;

    // Workspace (f16): Qh (8MB) | Kh (8MB) | attn (8MB) | yt (64MB)
    _Float16* Qh   = (_Float16*)d_ws;
    _Float16* Kh   = Qh + (size_t)BB * HH * WW;
    _Float16* attn = Kh + (size_t)BB * HH * WW;
    _Float16* yt   = attn + (size_t)BB * HH * HH;

    const size_t npix = (size_t)BB * HH * WW;          // 4M
    qk_proj_kernel<<<dim3((unsigned)(npix / 256)), dim3(256), 0, stream>>>(
        x, y, wq, bq, wk, bk, Qh, Kh);

    transpose_kernel<<<dim3(BB * CC * 32 * 32), dim3(256), 0, stream>>>(y, yt);

    attn_kernel<<<dim3(BB * (HH / 16)), dim3(256), 0, stream>>>(Qh, Kh, attn);

    out_kernel<<<dim3(BB * (HH / 32) * (WW / 64)), dim3(256), 0, stream>>>(
        attn, yt, x, wv, bv, gamma, outp);
}
